// TCNNSplitModel_16080357556230
// MI455X (gfx1250) — matrix operand order repaired and checked by reading, not compiled, not run on hardware
//
#include <hip/hip_runtime.h>
#include <hip/hip_bf16.h>

// ---------------------------------------------------------------------------
// Fused TCNN-style model for gfx1250 (MI455X, wave32, WMMA).
//   features: 24 triangle-wave + 4 hash-grid (lod-selected) + lod  (pad->32)
//   MLP:      [16x32]@[32x64] -> squareplus -> [16x64]@[64x64] -> squareplus
//             -> [16x64]@[64x16(pad from 3)]
// One wave = one 16-sample tile per grid-stride iteration.
// Weights live in VGPRs as WMMA B-fragments for the whole kernel.
// Layer-1 A-fragment is computed DIRECTLY in registers (no LDS round trip):
//   A layout: lane<16 : halves 0-7 = K 0..7,  halves 8-15 = K 16..23
//             lane>=16: halves 0-7 = K 8..15, halves 8-15 = K 24..31
//   K map: 0-11 tri(u,j), 12-23 tri(v,j), 24-27 hash feats, 28 lod, 29-31 pad.
// ---------------------------------------------------------------------------

typedef __attribute__((ext_vector_type(16))) _Float16 v16h;
typedef __attribute__((ext_vector_type(8)))  _Float16 v8h;
typedef __attribute__((ext_vector_type(8)))  float    v8f;

#define N_FREQ      12
#define TABLE_SIZE  (1u << 19)
#define HASH_PRIME  2654435761u
#define NNEUR       64

// fast squareplus: raw v_sqrt_f32 (1 ulp) -- feeding an f16 MLP, ample.
__device__ __forceinline__ float squareplus(float x) {
    return 0.5f * (x + __builtin_amdgcn_sqrtf(fmaf(x, x, 4.0f)));
}

// triangle wave: |x - floor(x+0.5)|*2 - 0.5 == |fract(x+0.5)-0.5|*2 - 0.5
__device__ __forceinline__ float triw(float x) {
    const float t = __builtin_amdgcn_fractf(x + 0.5f) - 0.5f;
    return fmaf(fabsf(t), 2.0f, -0.5f);
}

// Build a WMMA B-fragment (32-row K-block x 16-col tile) from row-major f32
// W[rows x strideCols], zero-padding rows >= rowsValid / cols >= colsValid.
// Dense-B layout: lanes 0-15 hold K = k0+0..7,16..23, lanes 16-31 hold K = k0+8..15,24..31,
// col = n0 + lane%16.
__device__ __forceinline__ v16h load_bfrag(const float* __restrict__ W,
                                           int rowsValid, int strideCols,
                                           int colsValid, int k0, int n0,
                                           int lane) {
    v16h b;
    const int kbase = k0 + ((lane >= 16) ? 8 : 0);
    const int col   = n0 + (lane & 15);
#pragma unroll
    for (int h = 0; h < 16; ++h) {
        const int k = kbase + h + ((h >= 8) ? 8 : 0);
        float v = 0.0f;
        if (k < rowsValid && col < colsValid)
            v = W[k * strideCols + col];
        b[h] = (_Float16)v;
    }
    return b;
}

// Read an A-fragment (16 rows x 32 K) from an LDS activation tile.
__device__ __forceinline__ v16h load_afrag(const _Float16* rowBase,
                                           int rowStrideH, int kOff,
                                           int lane) {
    const _Float16* r = rowBase + (lane & 15) * rowStrideH + kOff;
    const int klo = (lane >= 16) ? 8 : 0;
    v8h lo = *(const v8h*)(r + klo);        // 16B-aligned ds_load_b128
    v8h hi = *(const v8h*)(r + klo + 16);
    v16h a;
#pragma unroll
    for (int i = 0; i < 8; ++i) { a[i] = lo[i]; a[i + 8] = hi[i]; }
    return a;
}

// Bilinear hash-grid lookup for one level (tbl = level base, float2 entries).
__device__ __forceinline__ void hash2(const float* __restrict__ tbl,
                                      float u, float v, float res,
                                      float& f0, float& f1) {
    const float px = u * res, py = v * res;
    const float fx = floorf(px), fy = floorf(py);
    const float wx = px - fx, wy = py - fy;
    const unsigned ix = (unsigned)(int)fx, iy = (unsigned)(int)fy;
    f0 = 0.0f; f1 = 0.0f;
#pragma unroll
    for (int dy = 0; dy < 2; ++dy) {
#pragma unroll
        for (int dx = 0; dx < 2; ++dx) {
            const unsigned hx = ix + (unsigned)dx;
            const unsigned hy = iy + (unsigned)dy;
            const unsigned hh = (hx ^ (hy * HASH_PRIME)) & (TABLE_SIZE - 1u);
            const float wgt = (dx ? wx : 1.0f - wx) * (dy ? wy : 1.0f - wy);
            const float* e = tbl + (size_t)hh * 2u;
            f0 += wgt * e[0];
            f1 += wgt * e[1];
        }
    }
}

__device__ __forceinline__ void lds_fence_wave() {
    __builtin_amdgcn_wave_barrier();                    // sched barrier
    asm volatile("s_wait_dscnt 0x0" ::: "memory");      // LDS data visible
}

__global__ __launch_bounds__(256)
void tcnn_fused_kernel(const float* __restrict__ x,
                       const float* __restrict__ noise_p,
                       const float* __restrict__ hg0,
                       const float* __restrict__ hg1,
                       const float* __restrict__ hg2,
                       const float* __restrict__ w_in,
                       const float* __restrict__ w_h,
                       const float* __restrict__ w_out,
                       float* __restrict__ out,
                       int B) {
    __shared__ __align__(16) _Float16 s_h[8][16 * 64];  // per-wave activations

    const int tid  = threadIdx.x;
    const int lane = tid & 31;
    const int wid  = tid >> 5;
    const int waveGlobal = blockIdx.x * 8 + wid;
    const int nWaves     = gridDim.x * 8;
    const int nTiles     = (B + 15) >> 4;

    // ---- weight fragments, resident in VGPRs ----
    v16h bin[4];
#pragma unroll
    for (int n = 0; n < 4; ++n)
        bin[n] = load_bfrag(w_in, 29, NNEUR, NNEUR, 0, 16 * n, lane);
    v16h bh[2][4];
#pragma unroll
    for (int kb = 0; kb < 2; ++kb)
#pragma unroll
        for (int n = 0; n < 4; ++n)
            bh[kb][n] = load_bfrag(w_h, NNEUR, NNEUR, NNEUR, 32 * kb, 16 * n, lane);
    v16h bo[2];
#pragma unroll
    for (int kb = 0; kb < 2; ++kb)
        bo[kb] = load_bfrag(w_out, NNEUR, 3, 3, 32 * kb, 0, lane);

    const float noise = noise_p[0];
    _Float16* hbuf = &s_h[wid][0];

    const int row = lane & 15;       // sample within tile
    const bool hi = (lane >= 16);

    for (int tile = waveGlobal; tile < nTiles; tile += nWaves) {
        // prefetch next tile's inputs into cache (global_prefetch_b8)
        const int ntile = tile + nWaves;
        if (ntile < nTiles)
            __builtin_prefetch(&x[(size_t)(ntile * 16 + row) * 3], 0, 1);

        int s = tile * 16 + row;
        const int ss = (s < B) ? s : (B - 1);
        const float u   = x[(size_t)ss * 3 + 0];
        const float v   = x[(size_t)ss * 3 + 1];
        const float lod = x[(size_t)ss * 3 + 2];

        // ---- layer-1 A-fragment computed directly in registers ----
        v16h a_in;
        if (!hi) {
            // K = 0..7  -> tri(u, j=0..7)
#pragma unroll
            for (int h = 0; h < 8; ++h)
                a_in[h] = (_Float16)triw(u * (float)(1 << h));
            // K = 16..23 -> tri(v, j=4..11)
#pragma unroll
            for (int h = 8; h < 16; ++h)
                a_in[h] = (_Float16)triw(v * (float)(1 << (h - 4)));
        } else {
            // K = 8..11 -> tri(u, j=8..11)
#pragma unroll
            for (int h = 0; h < 4; ++h)
                a_in[h] = (_Float16)triw(u * (float)(1 << (h + 8)));
            // K = 12..15 -> tri(v, j=0..3)
#pragma unroll
            for (int h = 4; h < 8; ++h)
                a_in[h] = (_Float16)triw(v * (float)(1 << (h - 4)));
            // K = 24..27 -> hash features (levels 0 and 1), +noise
            const float* hg   = (lod == 0.0f) ? hg0 : ((lod == 1.0f) ? hg1 : hg2);
            const float  bres = (lod == 0.0f) ? 128.0f : ((lod == 1.0f) ? 32.0f : 8.0f);
            float f0, f1, f2, f3;
            hash2(hg, u, v, bres, f0, f1);
            hash2(hg + (size_t)TABLE_SIZE * 2u, u, v, 2.0f * bres, f2, f3);
            a_in[8]  = (_Float16)(f0 + noise);
            a_in[9]  = (_Float16)(f1 + noise);
            a_in[10] = (_Float16)(f2 + noise);
            a_in[11] = (_Float16)(f3 + noise);
            // K = 28 -> lod ; K = 29..31 -> pad
            a_in[12] = (_Float16)lod;
            a_in[13] = (_Float16)0.0f;
            a_in[14] = (_Float16)0.0f;
            a_in[15] = (_Float16)0.0f;
        }

        // ---- layer 1: [16x32] @ [32x64] ----
        v8f c1[4];
#pragma unroll
        for (int n = 0; n < 4; ++n) {
            v8f z = {};
            c1[n] = __builtin_amdgcn_wmma_f32_16x16x32_f16(
                        false, a_in, false, bin[n], (short)0, z, false, false);
        }
        // squareplus -> LDS [sample][neuron] f16
#pragma unroll
        for (int n = 0; n < 4; ++n) {
#pragma unroll
            for (int r = 0; r < 8; ++r) {
                const int M = r + (hi ? 8 : 0);
                const int N = 16 * n + row;
                hbuf[M * 64 + N] = (_Float16)squareplus(c1[n][r]);
            }
        }
        lds_fence_wave();

        // ---- layer 2: [16x64] @ [64x64], K split 2x32 ----
        const v16h a0 = load_afrag(hbuf, 64, 0, lane);
        const v16h a1 = load_afrag(hbuf, 64, 32, lane);
        v8f c2[4];
#pragma unroll
        for (int n = 0; n < 4; ++n) {
            v8f z = {};
            z = __builtin_amdgcn_wmma_f32_16x16x32_f16(
                    false, a0, false, bh[0][n], (short)0, z, false, false);
            c2[n] = __builtin_amdgcn_wmma_f32_16x16x32_f16(
                    false, a1, false, bh[1][n], (short)0, z, false, false);
        }
#pragma unroll
        for (int n = 0; n < 4; ++n) {
#pragma unroll
            for (int r = 0; r < 8; ++r) {
                const int M = r + (hi ? 8 : 0);
                const int N = 16 * n + row;
                hbuf[M * 64 + N] = (_Float16)squareplus(c2[n][r]);
            }
        }
        lds_fence_wave();

        // ---- layer 3: [16x64] @ [64x16(3 valid)] ----
        const v16h a2 = load_afrag(hbuf, 64, 0, lane);
        const v16h a3 = load_afrag(hbuf, 64, 32, lane);
        v8f co = {};
        co = __builtin_amdgcn_wmma_f32_16x16x32_f16(
                 false, a2, false, bo[0], (short)0, co, false, false);
        co = __builtin_amdgcn_wmma_f32_16x16x32_f16(
                 false, a3, false, bo[1], (short)0, co, false, false);

        // ---- store out[sample][0..2] ----
        const int N = row;
        if (N < 3) {
#pragma unroll
            for (int r = 0; r < 8; ++r) {
                const int M    = r + (hi ? 8 : 0);
                const int sidx = tile * 16 + M;
                if (sidx < B)
                    out[(size_t)sidx * 3 + N] = co[r];
            }
        }
    }
}

extern "C" void kernel_launch(void* const* d_in, const int* in_sizes, int n_in,
                              void* d_out, int out_size, void* d_ws, size_t ws_size,
                              hipStream_t stream) {
    const float* x     = (const float*)d_in[0];
    const float* noise = (const float*)d_in[1];
    const float* hg0   = (const float*)d_in[2];
    const float* hg1   = (const float*)d_in[3];
    const float* hg2   = (const float*)d_in[4];
    const float* w_in  = (const float*)d_in[5];
    const float* w_h   = (const float*)d_in[6];
    const float* w_out = (const float*)d_in[7];
    float* out = (float*)d_out;

    const int B = in_sizes[0] / 3;
    const int nTiles = (B + 15) / 16;
    int blocks = (nTiles + 7) / 8;          // 8 waves (tiles) per block
    if (blocks > 1024) blocks = 1024;       // grid-stride beyond this
    if (blocks < 1) blocks = 1;

    tcnn_fused_kernel<<<blocks, 256, 0, stream>>>(
        x, noise, hg0, hg1, hg2, w_in, w_h, w_out, out, B);
}
